// GCNEncoder_69810398429751
// MI455X (gfx1250) — matrix operand order repaired and checked by reading, not compiled, not run on hardware
//
#include <hip/hip_runtime.h>
#include <hip/hip_bf16.h>
#include <stdint.h>

// ---------------------------------------------------------------------------
// GCN 2-layer forward for gfx1250 (MI455X).
// Dense GEMMs: bf16 WMMA (v_wmma_f32_16x16x32_bf16), f32 accumulate,
//   with double-buffered GLOBAL_LOAD_ASYNC_TO_LDS_B128 staging (ASYNCcnt).
// Sparse aggregation: fp32 gather + global_atomic_add_f32 scatter.
// ---------------------------------------------------------------------------

typedef __attribute__((ext_vector_type(16))) __bf16 v16bf;
typedef __attribute__((ext_vector_type(8)))  float  v8f;

__device__ __forceinline__ unsigned short f2bf(float f) {
    union { float f; uint32_t u; } v; v.f = f;
    uint32_t u = v.u;
    uint32_t r = u + 0x7FFFu + ((u >> 16) & 1u);   // round-to-nearest-even
    return (unsigned short)(r >> 16);
}

// Async 16-byte global -> LDS copy (per-lane addresses), tracked by ASYNCcnt.
__device__ __forceinline__ void async_copy16(uint32_t lds_off, const void* gaddr) {
    asm volatile("global_load_async_to_lds_b128 %0, %1, off"
                 :: "v"(lds_off), "v"(gaddr) : "memory");
}
__device__ __forceinline__ void wait_async4() {
    asm volatile("s_wait_asynccnt 0x4" ::: "memory");
}
__device__ __forceinline__ void wait_async0() {
    asm volatile("s_wait_asynccnt 0x0" ::: "memory");
}

// ---------------- small elementwise kernels --------------------------------

__global__ void k_fill_const(float* __restrict__ p, float val, int n) {
    int i = blockIdx.x * 256 + threadIdx.x;
    if (i < n) p[i] = val;
}

__global__ void k_deg_count(const long long* __restrict__ ei, float* __restrict__ deg, int E) {
    int e = blockIdx.x * 256 + threadIdx.x;
    if (e < E) {
        int d = (int)ei[(size_t)E + e];          // dst row of edge_index
        atomicAdd(&deg[d], 1.0f);
    }
}

__global__ void k_rsqrt(float* __restrict__ d, int n) {
    int i = blockIdx.x * 256 + threadIdx.x;
    if (i < n) d[i] = rsqrtf(d[i]);              // deg >= 1 (self-loop)
}

__global__ void k_conv_bf16(const float4* __restrict__ in, uint2* __restrict__ out, int n4) {
    int i = blockIdx.x * 256 + threadIdx.x;
    if (i < n4) {
        float4 v = in[i];
        uint2 o;
        o.x = (uint32_t)f2bf(v.x) | ((uint32_t)f2bf(v.y) << 16);
        o.y = (uint32_t)f2bf(v.z) | ((uint32_t)f2bf(v.w) << 16);
        out[i] = o;
    }
}

// Wt[n*K + k] = bf16(W[k*N + n])   (pre-transpose weights to [N][K] bf16)
__global__ void k_transpose_w(const float* __restrict__ W, unsigned short* __restrict__ Wt,
                              int K, int N) {
    int i = blockIdx.x * 256 + threadIdx.x;
    if (i < K * N) {
        int n = i / K;
        int k = i - n * K;
        Wt[i] = f2bf(W[(size_t)k * N + n]);
    }
}

__global__ void k_fill_bias(float* __restrict__ out, const float* __restrict__ b,
                            int mask, int total) {
    int i = blockIdx.x * 256 + threadIdx.x;
    if (i < total) out[i] = b[i & mask];
}

// out_bf16 = bf16(relu(agg + gemm * dinv[row]^2))   (fused self-loop + ReLU)
__global__ void k_relu_selfloop_bf16(const float* __restrict__ agg,
                                     const float* __restrict__ gem,
                                     const float* __restrict__ dinv,
                                     unsigned short* __restrict__ hb,
                                     int shift, int total) {
    int i = blockIdx.x * 256 + threadIdx.x;
    if (i < total) {
        float di = dinv[i >> shift];
        float v  = agg[i] + gem[i] * di * di;
        hb[i] = f2bf(fmaxf(v, 0.0f));
    }
}

// out += gemm * dinv[row]^2   (final self-loop, fp32 output)
__global__ void k_selfloop_add(float* __restrict__ out,
                               const float* __restrict__ gem,
                               const float* __restrict__ dinv,
                               int shift, int total) {
    int i = blockIdx.x * 256 + threadIdx.x;
    if (i < total) {
        float di = dinv[i >> shift];
        out[i] += gem[i] * di * di;
    }
}

// ---------------- edge scatter-add (one wave per edge) ---------------------

__global__ __launch_bounds__(256) void k_edge_agg(
    const long long* __restrict__ ei, int E,
    const float* __restrict__ h, const float* __restrict__ dinv,
    float* __restrict__ out, int cols)
{
    int wid  = blockIdx.x * 8 + (threadIdx.x >> 5);
    int lane = threadIdx.x & 31;
    if (wid >= E) return;
    int s = (int)ei[wid];
    int d = (int)ei[(size_t)E + wid];
    float nrm = dinv[s] * dinv[d];
    const float4* hrow = (const float4*)(h + (size_t)s * cols);
    float*        orow = out + (size_t)d * cols;
    int c4n = cols >> 2;
    for (int c4 = lane; c4 < c4n; c4 += 32) {
        float4 v = hrow[c4];
        atomicAdd(orow + c4 * 4 + 0, v.x * nrm);
        atomicAdd(orow + c4 * 4 + 1, v.y * nrm);
        atomicAdd(orow + c4 * 4 + 2, v.z * nrm);
        atomicAdd(orow + c4 * 4 + 3, v.w * nrm);
    }
}

// ---------------- bf16 WMMA GEMM:  C[M,N] = A[M,K] * B^T  (B stored [N][K]) -

#define BM 64
#define BN 64
#define BKT 32
#define LPITCH 40   // bf16 elements per LDS row (80 B: 16B-aligned, conflict-free)

__global__ __launch_bounds__(128) void k_gemm_bf16(
    const unsigned short* __restrict__ A,   // [M][K] bf16
    const unsigned short* __restrict__ B,   // [N][K] bf16 (transposed weights)
    float* __restrict__ C,                  // [M][N] f32
    int M, int K, int N)
{
    __shared__ alignas(16) unsigned short lA[2][BM * LPITCH];
    __shared__ alignas(16) unsigned short lB[2][BN * LPITCH];

    const int tid  = threadIdx.x;
    const int wave = tid >> 5;
    const int lane = tid & 31;
    const int l16  = lane & 15;
    const int half = lane >> 4;

    const int m0 = blockIdx.y * BM;
    const int n0 = blockIdx.x * BN;

    // per-thread staging coordinates (2 chunks of 8 bf16 for A, 2 for B)
    int r0   = tid >> 2;                  // 0..31
    int r1   = (tid + 128) >> 2;          // 32..63
    int seg0 = tid & 3;
    int seg1 = (tid + 128) & 3;
    int ga0  = m0 + r0; if (ga0 >= M) ga0 = M - 1;   // clamp (discarded at store)
    int ga1  = m0 + r1; if (ga1 >= M) ga1 = M - 1;

    // issue the 4 async b128 copies for tile (buf, k0) of this thread
    auto stage = [&](int buf, int k0) {
        async_copy16((uint32_t)(uintptr_t)&lA[buf][r0 * LPITCH + seg0 * 8],
                     A + (size_t)ga0 * K + k0 + seg0 * 8);
        async_copy16((uint32_t)(uintptr_t)&lA[buf][r1 * LPITCH + seg1 * 8],
                     A + (size_t)ga1 * K + k0 + seg1 * 8);
        async_copy16((uint32_t)(uintptr_t)&lB[buf][r0 * LPITCH + seg0 * 8],
                     B + (size_t)(n0 + r0) * K + k0 + seg0 * 8);
        async_copy16((uint32_t)(uintptr_t)&lB[buf][r1 * LPITCH + seg1 * 8],
                     B + (size_t)(n0 + r1) * K + k0 + seg1 * 8);
    };

    v8f acc[4] = {};

    const int nk = K / BKT;
    stage(0, 0);

    for (int i = 0; i < nk; ++i) {
        const int buf = i & 1;
        if (i + 1 < nk) {
            stage(buf ^ 1, (i + 1) * BKT);   // prefetch next tile (4 more async ops)
            wait_async4();                   // oldest 4 (current tile) landed
        } else {
            wait_async0();
        }
        __syncthreads();

        // A fragment: lane element e -> K = (e<8?0:16) + half*8 + (e%8), M = wave*16 + l16
        union { uint4 q[2]; v16bf v; } af;
        {
            const char* base = (const char*)&lA[buf][0] + (wave * 16 + l16) * (LPITCH * 2);
            af.q[0] = *(const uint4*)(base + half * 16);
            af.q[1] = *(const uint4*)(base + 32 + half * 16);
        }

        for (int t = 0; t < 4; ++t) {
            // B fragment: lane element e -> K = (e<8?0:16) + half*8 + (e%8), N = t*16 + l16
            union { uint4 q[2]; v16bf v; } bf;
            const char* base = (const char*)&lB[buf][0] + (t * 16 + l16) * (LPITCH * 2);
            bf.q[0] = *(const uint4*)(base + half * 16);
            bf.q[1] = *(const uint4*)(base + 32 + half * 16);

            acc[t] = __builtin_amdgcn_wmma_f32_16x16x32_bf16(
                /*neg_a=*/false, af.v, /*neg_b=*/false, bf.v,
                /*c_mod=*/(short)0, acc[t], /*reuse_a=*/false, /*reuse_b=*/false);
        }
        __syncthreads();   // protect buf from being re-staged while others still read
    }

    // C/D layout: VGPR r -> row = half*8 + r, col = lane%16
    for (int t = 0; t < 4; ++t) {
        int col = n0 + t * 16 + l16;
        for (int r = 0; r < 8; ++r) {
            int row = m0 + wave * 16 + half * 8 + r;
            if (row < M) C[(size_t)row * N + col] = acc[t][r];
        }
    }
}

// ---------------------------------------------------------------------------

extern "C" void kernel_launch(void* const* d_in, const int* in_sizes, int n_in,
                              void* d_out, int out_size, void* d_ws, size_t ws_size,
                              hipStream_t stream) {
    const float*     x  = (const float*)d_in[0];
    const long long* ei = (const long long*)d_in[1];
    const float*     W1 = (const float*)d_in[2];
    const float*     b1 = (const float*)d_in[3];
    const float*     W2 = (const float*)d_in[4];
    const float*     b2 = (const float*)d_in[5];

    const int hid    = in_sizes[3];            // 256
    const int outd   = in_sizes[5];            // 128
    const int in_dim = in_sizes[2] / hid;      // 128
    const int N      = in_sizes[0] / in_dim;   // 50000
    const int E      = in_sizes[1] / 2;        // 800000

    // workspace layout (256B aligned slices)
    size_t off = 0;
    auto take = [&](size_t bytes) { size_t o = off; off += (bytes + 255) & ~(size_t)255; return o; };
    float*          dinv  = (float*)((char*)d_ws + take((size_t)N * 4));
    unsigned short* xb    = (unsigned short*)((char*)d_ws + take((size_t)N * in_dim * 2));
    unsigned short* wt1   = (unsigned short*)((char*)d_ws + take((size_t)in_dim * hid * 2));
    unsigned short* wt2   = (unsigned short*)((char*)d_ws + take((size_t)hid * outd * 2));
    float*          gemmO = (float*)((char*)d_ws + take((size_t)N * hid * 4));
    float*          aggO  = (float*)((char*)d_ws + take((size_t)N * hid * 4));
    unsigned short* h1b   = (unsigned short*)((char*)d_ws + take((size_t)N * hid * 2));
    (void)ws_size;

    const int shiftH = 31 - __builtin_clz(hid);    // log2(256) = 8
    const int shiftO = 31 - __builtin_clz(outd);   // log2(128) = 7

    // --- degree / normalization ---
    k_fill_const<<<(N + 255) / 256, 256, 0, stream>>>(dinv, 1.0f, N); // self-loop
    k_deg_count <<<(E + 255) / 256, 256, 0, stream>>>(ei, dinv, E);
    k_rsqrt     <<<(N + 255) / 256, 256, 0, stream>>>(dinv, N);

    // --- precision conversion ---
    {
        int n4 = (N * in_dim) / 4;
        k_conv_bf16<<<(n4 + 255) / 256, 256, 0, stream>>>((const float4*)x, (uint2*)xb, n4);
    }
    k_transpose_w<<<(in_dim * hid + 255) / 256, 256, 0, stream>>>(W1, wt1, in_dim, hid);
    k_transpose_w<<<(hid * outd + 255) / 256, 256, 0, stream>>>(W2, wt2, hid, outd);

    // --- layer 1: GEMM -> scatter -> (self-loop + bias + relu) -> bf16 ---
    {
        dim3 g(hid / BN, (N + BM - 1) / BM);
        k_gemm_bf16<<<g, 128, 0, stream>>>(xb, wt1, gemmO, N, in_dim, hid);
    }
    {
        int total = N * hid;
        k_fill_bias<<<(total + 255) / 256, 256, 0, stream>>>(aggO, b1, hid - 1, total);
        k_edge_agg <<<(E + 7) / 8, 256, 0, stream>>>(ei, E, gemmO, dinv, aggO, hid);
        k_relu_selfloop_bf16<<<(total + 255) / 256, 256, 0, stream>>>(
            aggO, gemmO, dinv, h1b, shiftH, total);
    }

    // --- layer 2: GEMM -> scatter into d_out -> self-loop ---
    {
        dim3 g(outd / BN, (N + BM - 1) / BM);
        k_gemm_bf16<<<g, 128, 0, stream>>>(h1b, wt2, gemmO, N, hid, outd);
    }
    {
        float* out = (float*)d_out;
        int total  = N * outd;
        k_fill_bias<<<(total + 255) / 256, 256, 0, stream>>>(out, b2, outd - 1, total);
        k_edge_agg <<<(E + 7) / 8, 256, 0, stream>>>(ei, E, gemmO, dinv, out, outd);
        k_selfloop_add<<<(total + 255) / 256, 256, 0, stream>>>(out, gemmO, dinv, shiftO, total);
    }
}
